// Attention_49718541418817
// MI455X (gfx1250) — matrix operand order repaired and checked by reading, not compiled, not run on hardware
//
#include <hip/hip_runtime.h>
#include <math.h>

#define DEV __device__ __forceinline__

typedef _Float16 v16h __attribute__((ext_vector_type(16)));
typedef _Float16 v8h  __attribute__((ext_vector_type(8)));
typedef float    v8f  __attribute__((ext_vector_type(8)));
typedef float    v4f  __attribute__((ext_vector_type(4)));
typedef unsigned int v4u __attribute__((ext_vector_type(4)));
typedef int      v8i  __attribute__((ext_vector_type(8)));
typedef int      v4i  __attribute__((ext_vector_type(4)));

static constexpr int NB = 8;        // batch
static constexpr int NN = 4096;     // H*W tokens
static constexpr int NC = 256;      // channels
static constexpr int ND = 32;       // head dim
static constexpr int ROWS = NB * NN;

// workspace layout (bytes); total ~6.3 MB
static constexpr size_t OFF_KF = 0;                                     // [32][256] f16 (kf^T / sigma)
static constexpr size_t OFF_KG = OFF_KF + (size_t)ND * NC * 2;
static constexpr size_t OFF_KH = OFF_KG + (size_t)ND * NC * 2;
static constexpr size_t OFF_KV = OFF_KH + (size_t)ND * NC * 2;          // [256][32] f16 (kv^T / sigma)
static constexpr size_t OFF_F  = OFF_KV + (size_t)NC * ND * 2;          // [32768][32] f16
static constexpr size_t OFF_G  = OFF_F + (size_t)ROWS * ND * 2;         // [32768][32] f16
static constexpr size_t OFF_HT = OFF_G + (size_t)ROWS * ND * 2;         // [8][32][4096] f16

DEV v8f zero8() { v8f z = {0.f,0.f,0.f,0.f,0.f,0.f,0.f,0.f}; return z; }

DEV v8f wmma16(v16h a, v16h b, v8f c) {
  return __builtin_amdgcn_wmma_f32_16x16x32_f16(false, a, false, b, (short)0, c, false, false);
}

// A-operand load: two 16B chunks (K{0..7}/{8..15} or K{16..23}/{24..31})
DEV v16h ld_a(const _Float16* p0, const _Float16* p1) {
  union { v16h v; v8h h[2]; } u;
  u.h[0] = *(const v8h*)p0;
  u.h[1] = *(const v8h*)p1;
  return u.v;
}
// B-operand load: 32 contiguous bytes (16 consecutive K values for this lane half)
DEV v16h ld_b(const _Float16* p) {
  union { v16h v; v8h h[2]; } u;
  u.h[0] = *(const v8h*)p;
  u.h[1] = *(const v8h*)(p + 8);
  return u.v;
}
// 8 f32 -> 8 f16
DEV v8h cvt8(const float* p) {
  v4f a = *(const v4f*)p;
  v4f b = *(const v4f*)(p + 4);
  v8h r;
#pragma unroll
  for (int i = 0; i < 4; ++i) { r[i] = (_Float16)a[i]; r[4 + i] = (_Float16)b[i]; }
  return r;
}

DEV float red16_max(float v) {
#pragma unroll
  for (int m = 1; m < 16; m <<= 1) v = fmaxf(v, __shfl_xor(v, m, 32));
  return v;
}
DEV float red16_sum(float v) {
#pragma unroll
  for (int m = 1; m < 16; m <<= 1) v += __shfl_xor(v, m, 32);
  return v;
}

DEV unsigned lds_off(const void* p) { return (unsigned)(size_t)p; }

// TDM 2-D tile load: 32 rows x 32 f16 (64B) per row, row pitch `stride0`
// elements, gathered compact into LDS at lds_addr. D# per CDNA5 ISA:
//  group0: count=1 | lds_addr | global_addr[31:0] | global_addr[56:32], type=2
//  group1: data_size=1(2B); tensor_dim0=32; tensor_dim1=32; tile_dim0=32;
//          tile_dim1=32; tensor_dim0_stride=stride0
DEV void tdm_issue_32x32(unsigned lds_addr, const void* gptr, unsigned stride0) {
  unsigned long long ga = (unsigned long long)(size_t)gptr;
  v4u g0 = { 1u, lds_addr, (unsigned)ga,
             (unsigned)((ga >> 32) & 0x01ffffffu) | 0x80000000u };
  v8i g1 = { (int)0x00010000,      // wg_mask=0, data_size=1 (2 bytes)
             (int)0x00200000,      // atomic_addr=0 | tensor_dim0(=32) low16
             (int)0x00200000,      // tensor_dim0 hi16=0 | tensor_dim1(=32) low16
             (int)0x00200000,      // tensor_dim1 hi16=0 | tile_dim0=32
             32,                   // tile_dim1=32, tile_dim2=0
             (int)stride0,         // tensor_dim0_stride low32
             0, 0 };               // stride hi / tensor_dim1_stride (unused, 2D)
  v4i z4 = { 0, 0, 0, 0 };
  v8i z8 = { 0, 0, 0, 0, 0, 0, 0, 0 };
  __builtin_amdgcn_tensor_load_to_lds(g0, g1, z4, z4, z8, 0);
}

// ---------------------------------------------------------------------------
// Kernel 1: spectral norm (sigma = ||l2n(u W^T) W||) + transposed f16 weights
// ---------------------------------------------------------------------------
__global__ void __launch_bounds__(256) sn_normalize_kernel(
    const float* __restrict__ wf, const float* __restrict__ uf,
    const float* __restrict__ wg, const float* __restrict__ ug,
    const float* __restrict__ wh, const float* __restrict__ uh,
    const float* __restrict__ wvw, const float* __restrict__ uvw,
    _Float16* __restrict__ kf, _Float16* __restrict__ kg,
    _Float16* __restrict__ kh, _Float16* __restrict__ kv)
{
  const float* w; const float* u; _Float16* o; int I, O;
  switch (blockIdx.x) {
    case 0:  w = wf;  u = uf;  o = kf; I = NC; O = ND; break;
    case 1:  w = wg;  u = ug;  o = kg; I = NC; O = ND; break;
    case 2:  w = wh;  u = uh;  o = kh; I = NC; O = ND; break;
    default: w = wvw; u = uvw; o = kv; I = ND; O = NC; break;
  }
  __shared__ float sv[256];
  __shared__ float red[256];
  const int tid = threadIdx.x;

  float vi = 0.f;
  if (tid < I) {
    for (int j = 0; j < O; ++j) vi += u[j] * w[tid * O + j];
  }
  red[tid] = vi * vi;
  __syncthreads();
  for (int s = 128; s > 0; s >>= 1) { if (tid < s) red[tid] += red[tid + s]; __syncthreads(); }
  float inv = 1.0f / sqrtf(fmaxf(red[0], 1e-12f));
  __syncthreads();
  if (tid < I) sv[tid] = vi * inv;
  __syncthreads();

  float tj = 0.f;
  if (tid < O) {
    for (int i = 0; i < I; ++i) tj += sv[i] * w[i * O + tid];
  }
  red[tid] = tj * tj;
  __syncthreads();
  for (int s = 128; s > 0; s >>= 1) { if (tid < s) red[tid] += red[tid + s]; __syncthreads(); }
  float ssq = red[0];
  float sigma = ssq / sqrtf(fmaxf(ssq, 1e-12f));
  float invs = 1.0f / sigma;

  for (int idx = tid; idx < I * O; idx += 256) {
    int i = idx / O, j = idx - i * O;
    o[j * I + i] = (_Float16)(w[idx] * invs);
  }
}

// ---------------------------------------------------------------------------
// Kernel 2: projections f/g/h = x @ k + b via WMMA. One wave per 16-row tile.
// ---------------------------------------------------------------------------
__global__ void __launch_bounds__(128) proj_kernel(
    const float* __restrict__ x,
    const _Float16* __restrict__ kf, const _Float16* __restrict__ kg,
    const _Float16* __restrict__ kh,
    const float* __restrict__ bf, const float* __restrict__ bg,
    const float* __restrict__ bh,
    _Float16* __restrict__ fbuf, _Float16* __restrict__ gbuf,
    _Float16* __restrict__ hT)
{
  const int lane = threadIdx.x & 31;
  const int wid  = threadIdx.x >> 5;
  const int tile = blockIdx.x * 4 + wid;
  const int rbase = tile * 16;
  const int lh = lane & 15, hi = lane >> 4;

  const float* xr = x + (size_t)(rbase + lh) * NC;

  v8f accF[2] = { zero8(), zero8() };
  v8f accG[2] = { zero8(), zero8() };
  v8f accH[2] = { zero8(), zero8() };

#pragma unroll
  for (int kc = 0; kc < 8; ++kc) {
    const int c0 = kc * 32;
    v16h a;
    {
      union { v16h v; v8h h[2]; } ua;
      ua.h[0] = cvt8(xr + c0 + hi * 16);
      ua.h[1] = cvt8(xr + c0 + hi * 16 + 8);
      a = ua.v;
    }
    const int bofs = c0 + hi * 16;
#pragma unroll
    for (int jt = 0; jt < 2; ++jt) {
      const int jrow = (jt * 16 + lh) * NC + bofs;
      v16h bF = ld_b(kf + jrow);
      v16h bG = ld_b(kg + jrow);
      v16h bH = ld_b(kh + jrow);
      accF[jt] = wmma16(a, bF, accF[jt]);
      accG[jt] = wmma16(a, bG, accG[jt]);
      accH[jt] = wmma16(a, bH, accH[jt]);
    }
  }

#pragma unroll
  for (int jt = 0; jt < 2; ++jt) {
    const int j = jt * 16 + lh;
    const float bfv = bf[j], bgv = bg[j], bhv = bh[j];
#pragma unroll
    for (int r = 0; r < 8; ++r) {
      const int row = rbase + r + 8 * hi;
      fbuf[row * ND + j] = (_Float16)(accF[jt][r] + bfv);
      gbuf[row * ND + j] = (_Float16)(accG[jt][r] + bgv);
      const int bb = row >> 12;
      const int n  = row & (NN - 1);
      hT[((size_t)(bb * ND + j)) * NN + n] = (_Float16)(accH[jt][r] + bhv);
    }
  }
}

// ---------------------------------------------------------------------------
// Kernel 3: fused flash attention + output projection + residual.
// 8 waves/block (8 query tiles, same batch). Key/value 32x32 f16 tiles are
// DMA'd into LDS by the Tensor Data Mover (double buffered); all 8 waves
// consume them via ds_load_b128 WMMA-B operands.
// ---------------------------------------------------------------------------
__global__ void __launch_bounds__(256) attn_kernel(
    const _Float16* __restrict__ gbuf, const _Float16* __restrict__ fbuf,
    const _Float16* __restrict__ hT,   const _Float16* __restrict__ kvT,
    const float* __restrict__ bv, const float* __restrict__ gamma,
    const float* __restrict__ x, float* __restrict__ out)
{
  __shared__ _Float16 sf[2][32 * 32];     // staged keys   f[kb+0..31][0..31]
  __shared__ _Float16 sh[2][32 * 32];     // staged values hT[dim 0..31][kb+0..31]
  __shared__ _Float16 plds[8][16 * 32];   // per-wave P-tile transpose buffer

  const int lane = threadIdx.x & 31;
  const int wid  = threadIdx.x >> 5;
  const int tile = blockIdx.x * 8 + wid;
  const int b  = tile >> 8;                // 256 tiles per batch
  const int q0 = (tile & 255) * 16;
  const int lh = lane & 15, hi = lane >> 4;
  _Float16* pl = &plds[wid][0];

  // queries (A operand), resident for the whole loop
  const _Float16* grow = gbuf + (size_t)(b * NN + q0 + lh) * ND;
  const v16h gA = ld_a(grow + hi * 16, grow + hi * 16 + 8);

  float m[8], l[8];
#pragma unroll
  for (int r = 0; r < 8; ++r) { m[r] = -1e30f; l[r] = 0.f; }
  v8f acc0 = zero8(), acc1 = zero8();

  const _Float16* fb = fbuf + (size_t)b * NN * ND;
  const _Float16* hb = hT   + (size_t)b * ND * NN;

  const unsigned sf_lds[2] = { lds_off(&sf[0][0]), lds_off(&sf[1][0]) };
  const unsigned sh_lds[2] = { lds_off(&sh[0][0]), lds_off(&sh[1][0]) };

  constexpr int ITERS = NN / 32;          // 128

  if (wid == 0) {                         // prologue: fill buffer 0
    tdm_issue_32x32(sf_lds[0], fb, ND);
    tdm_issue_32x32(sh_lds[0], hb, NN);
  }

  for (int i = 0; i < ITERS; ++i) {
    const int cur = i & 1;
    if (wid == 0) {
      if (i + 1 < ITERS) {                // prefetch next block via TDM
        const int kn = (i + 1) * 32;
        tdm_issue_32x32(sf_lds[1 - cur], fb + (size_t)kn * ND, ND);
        tdm_issue_32x32(sh_lds[1 - cur], hb + kn, NN);
        __builtin_amdgcn_s_wait_tensorcnt(2);   // retire the 2 loads for `cur`
      } else {
        __builtin_amdgcn_s_wait_tensorcnt(0);
      }
    }
    __syncthreads();                      // buffer `cur` is ready

    // keys (B operand): s = g . f^T
    v16h fB0 = ld_b(&sf[cur][lh * 32 + hi * 16]);
    v16h fB1 = ld_b(&sf[cur][(16 + lh) * 32 + hi * 16]);
    v8f s0 = wmma16(gA, fB0, zero8());
    v8f s1 = wmma16(gA, fB1, zero8());
    // values (B operand), dims 0-15 and 16-31
    v16h hB0 = ld_b(&sh[cur][lh * 32 + hi * 16]);
    v16h hB1 = ld_b(&sh[cur][(16 + lh) * 32 + hi * 16]);

    // online softmax per row (rows r+8*hi live in 16-lane groups)
#pragma unroll
    for (int r = 0; r < 8; ++r) {
      float s0r = s0[r], s1r = s1[r];
      float mnew = fmaxf(m[r], red16_max(fmaxf(s0r, s1r)));
      float sc = __expf(m[r] - mnew);
      float p0 = __expf(s0r - mnew);
      float p1 = __expf(s1r - mnew);
      float rs = red16_sum(p0 + p1);
      l[r] = l[r] * sc + rs;
      acc0[r] *= sc; acc1[r] *= sc;
      m[r] = mnew;
      pl[(r + 8 * hi) * 32 + lh]      = (_Float16)p0;
      pl[(r + 8 * hi) * 32 + 16 + lh] = (_Float16)p1;
    }
    // D-layout -> A-layout transpose through per-wave LDS (in-order DS + wait)
    asm volatile("s_wait_dscnt 0x0" ::: "memory");
    v16h pA = ld_a(pl + lh * 32 + hi * 16, pl + lh * 32 + hi * 16 + 8);
    acc0 = wmma16(pA, hB0, acc0);
    acc1 = wmma16(pA, hB1, acc1);

    __syncthreads();                      // done reading `cur` before overwrite
  }

  // normalize and stage attended values for the output projection
#pragma unroll
  for (int r = 0; r < 8; ++r) {
    float inv = 1.0f / l[r];
    acc0[r] *= inv; acc1[r] *= inv;
    pl[(r + 8 * hi) * 32 + lh]      = (_Float16)acc0[r];
    pl[(r + 8 * hi) * 32 + 16 + lh] = (_Float16)acc1[r];
  }
  asm volatile("s_wait_dscnt 0x0" ::: "memory");
  const v16h vA = ld_a(pl + lh * 32 + hi * 16, pl + lh * 32 + hi * 16 + 8);

  const float gm = gamma[0];
#pragma unroll 4
  for (int ct = 0; ct < 16; ++ct) {
    v16h kB = ld_b(kvT + (size_t)(ct * 16 + lh) * ND + hi * 16);
    v8f o = wmma16(vA, kB, zero8());
    const int c = ct * 16 + lh;
    const float bvv = bv[c];
#pragma unroll
    for (int r = 0; r < 8; ++r) {
      const int row = q0 + r + 8 * hi;
      const size_t idx = ((size_t)(b * NN + row)) * NC + c;
      out[idx] = gm * (o[r] + bvv) + x[idx];
    }
  }
}

// ---------------------------------------------------------------------------
extern "C" void kernel_launch(void* const* d_in, const int* in_sizes, int n_in,
                              void* d_out, int out_size, void* d_ws, size_t ws_size,
                              hipStream_t stream) {
  const float* x   = (const float*)d_in[0];
  const float* wf  = (const float*)d_in[1];
  const float* bf  = (const float*)d_in[2];
  const float* uf  = (const float*)d_in[3];
  const float* wg  = (const float*)d_in[4];
  const float* bg  = (const float*)d_in[5];
  const float* ug  = (const float*)d_in[6];
  const float* wh  = (const float*)d_in[7];
  const float* bh  = (const float*)d_in[8];
  const float* uh  = (const float*)d_in[9];
  const float* wvw = (const float*)d_in[10];
  const float* bv  = (const float*)d_in[11];
  const float* uvw = (const float*)d_in[12];
  const float* gamma = (const float*)d_in[13];
  float* out = (float*)d_out;

  char* ws = (char*)d_ws;
  _Float16* kf   = (_Float16*)(ws + OFF_KF);
  _Float16* kg   = (_Float16*)(ws + OFF_KG);
  _Float16* kh   = (_Float16*)(ws + OFF_KH);
  _Float16* kv   = (_Float16*)(ws + OFF_KV);
  _Float16* fbuf = (_Float16*)(ws + OFF_F);
  _Float16* gbuf = (_Float16*)(ws + OFF_G);
  _Float16* hT   = (_Float16*)(ws + OFF_HT);

  sn_normalize_kernel<<<4, 256, 0, stream>>>(wf, uf, wg, ug, wh, uh, wvw, uvw,
                                             kf, kg, kh, kv);
  proj_kernel<<<ROWS / 16 / 4, 128, 0, stream>>>(x, kf, kg, kh, bf, bg, bh,
                                                 fbuf, gbuf, hT);
  attn_kernel<<<ROWS / 16 / 8, 256, 0, stream>>>(gbuf, fbuf, hT, kv, bv, gamma,
                                                 x, out);
}
